// DDGNN_42279658062556
// MI455X (gfx1250) — matrix operand order repaired and checked by reading, not compiled, not run on hardware
//
#include <hip/hip_runtime.h>
#include <hip/hip_bf16.h>
#include <math.h>

// ---------------------------------------------------------------------------
// DDGNN for MI455X (gfx1250, wave32, WMMA bf16, TDM/async-LDS staging)
//   1) time-decay collapse  x[B,L,N,3] (+node_emb) -> x_adj [B,N,16(pad)]
//   2) DDL: m1,m2 = elu(x_adj@W+b)+E  -> bf16 [B,N,64]
//   3) S = m1 @ m2^T (WMMA bf16) -> S row-major + ST transposed copy (f32)
//   4) adj = softmax(tanh(S+S^T)/0.7): both operands now row-major streams
//      softmax rows sum to 1 => na = (adj+I)/2 exactly (no degree pass)
//   5) TCN: per-(b,n) sequence LDS-resident, gated dilated convs as
//      16x16x32 bf16 WMMA (M=oc, N=t, K=ic; 3 dilation-shifted B tiles)
//   6) APPNP x8: z' = a*xt + (1-a)*(0.5*adj@z + 0.5*z); z panel (64KB/batch)
//      staged to LDS via TDM / async-LDS; each wave does both c-tiles with
//      one A-frag so adj streams from HBM exactly once per iteration
//   7) head MLP 32->32->12
// ---------------------------------------------------------------------------

typedef __bf16 bf16;
typedef __attribute__((ext_vector_type(16))) __bf16 v16bf;
typedef __attribute__((ext_vector_type(8)))  float  v8f;
typedef __attribute__((ext_vector_type(4)))  unsigned int v4u;
typedef __attribute__((ext_vector_type(8)))  int v8i;
typedef __attribute__((ext_vector_type(4)))  int v4i;
typedef __attribute__((address_space(1))) void as1void;
typedef __attribute__((address_space(3))) void as3void;

#define NB   16
#define NL   96
#define NN   1024
#define NC   13
#define NH   64
#define NDC  32
#define NOUT 12
#define F_ALPHA 0.2f
#define F_DECAY 0.5f
#define F_TEMP  0.7f

#if __has_builtin(__builtin_amdgcn_tensor_load_to_lds)
#define USE_TDM 1
#else
#define USE_TDM 0
#endif
#if __has_builtin(__builtin_amdgcn_global_load_async_to_lds_b128)
#define USE_ASYNC_LDS 1
#else
#define USE_ASYNC_LDS 0
#endif

static __device__ __forceinline__ v8f wmma_bf16(v16bf a, v16bf b, v8f c) {
  return __builtin_amdgcn_wmma_f32_16x16x32_bf16(false, a, false, b, (short)0,
                                                 c, false, false);
}

// ---------------- 1) time-decay collapse ------------------------------------
__global__ void xadj_kernel(const float* __restrict__ x,
                            const float* __restrict__ ne,
                            float* __restrict__ xadj) {
  int idx = blockIdx.x * 256 + threadIdx.x;            // b*1024 + n
  if (idx >= NB * NN) return;
  int b = idx >> 10, n = idx & (NN - 1);
  const float r    = __expf(-F_DECAY);
  const float norm = (1.f - __powf(r, (float)NL)) / (1.f - r);
  float w = __powf(r, (float)(NL - 1));                // weight at l=0
  const float rinv = __expf(F_DECAY);
  float a0 = 0.f, a1 = 0.f, a2 = 0.f;
  const float* xp = x + (((size_t)b * NL) * NN + n) * 3;
  for (int l = 0; l < NL; ++l) {
    a0 += w * xp[0]; a1 += w * xp[1]; a2 += w * xp[2];
    xp += (size_t)NN * 3;
    w *= rinv;
  }
  float* o = xadj + (size_t)idx * 16;
  o[0] = a0 / norm; o[1] = a1 / norm; o[2] = a2 / norm;
#pragma unroll
  for (int e = 0; e < 10; ++e) o[3 + e] = ne[n * 10 + e];
  o[13] = 0.f; o[14] = 0.f; o[15] = 0.f;
}

// ---------------- 2) DDL ----------------------------------------------------
__global__ void ddl_kernel(const float* __restrict__ xadj,
                           const float* __restrict__ w1, const float* __restrict__ b1,
                           const float* __restrict__ w2, const float* __restrict__ b2,
                           const float* __restrict__ e1, const float* __restrict__ e2,
                           bf16* __restrict__ m1b, bf16* __restrict__ m2b) {
  int g    = threadIdx.x >> 6;                         // 4 nodes per block
  int h    = threadIdx.x & 63;
  int node = blockIdx.x * 4 + g;
  int n    = node & (NN - 1);
  __shared__ float sx[4][16];
  if (h < 16) sx[g][h] = xadj[(size_t)node * 16 + h];
  __syncthreads();
  float d1 = b1[h], d2 = b2[h];
#pragma unroll
  for (int i = 0; i < NC; ++i) {
    float xv = sx[g][i];
    d1 += xv * w1[i * NH + h];
    d2 += xv * w2[i * NH + h];
  }
  d1 = (d1 > 0.f) ? d1 : (__expf(d1) - 1.f);           // elu
  d2 = (d2 > 0.f) ? d2 : (__expf(d2) - 1.f);
  d1 += e1[n * NH + h];
  d2 += e2[n * NH + h];
  m1b[(size_t)node * NH + h] = (bf16)d1;
  m2b[(size_t)node * NH + h] = (bf16)d2;
}

// ---------------- 3) S = m1 @ m2^T  (WMMA) + transposed copy ----------------
__global__ __launch_bounds__(256)
void sgemm_kernel(const bf16* __restrict__ m1b,
                  const bf16* __restrict__ m2b,
                  float* __restrict__ S,
                  float* __restrict__ ST) {
  int wid  = blockIdx.x * 8 + (threadIdx.x >> 5);      // one 16x16 tile / wave
  int lane = threadIdx.x & 31;
  int b     = wid >> 12;
  int r     = wid & 4095;
  int ntile = r >> 6, mtile = r & 63;
  const bf16* m1 = m1b + ((size_t)b << 10) * NH;
  const bf16* m2 = m2b + ((size_t)b << 10) * NH;
  int nrow  = ntile * 16 + (lane & 15);
  int mrow  = mtile * 16 + (lane & 15);
  int khalf = (lane & 16) ? 8 : 0;
  int ksec  = (lane & 16) ? 16 : 0;
  v8f acc = {};
#pragma unroll
  for (int kc = 0; kc < 2; ++kc) {
    int kb = kc * 32;
    v16bf a, bb;
    const bf16* ar = m1 + (size_t)nrow * NH + kb + ksec;
#pragma unroll
    for (int i = 0; i < 8; ++i) { a[i] = ar[i]; a[8 + i] = ar[8 + i]; }
    const bf16* br = m2 + (size_t)mrow * NH + kb + ksec;
#pragma unroll
    for (int i = 0; i < 16; ++i) bb[i] = br[i];
    acc = wmma_bf16(a, bb, acc);
  }
  int m  = mtile * 16 + (lane & 15);
  int n0 = ntile * 16 + ((lane & 16) ? 8 : 0);
  float* Sp  = S  + ((size_t)b << 20);
  float* STp = ST + ((size_t)b << 20);
#pragma unroll
  for (int j = 0; j < 8; ++j) Sp[(size_t)(n0 + j) * NN + m] = acc[j];
  // transposed copy: ST[m][n] = S[n][m]; contiguous 8 floats per lane
#pragma unroll
  for (int j = 0; j < 8; ++j) STp[(size_t)m * NN + n0 + j] = acc[j];
}

// ---------------- 4) adj = softmax(tanh(S+S^T)/T) ---------------------------
// tanh/TEMP is bounded in [-1.43,1.43] so exp needs no max-subtraction.
__global__ void adj_kernel(const float* __restrict__ S,
                           const float* __restrict__ ST,
                           float* __restrict__ adj_out,
                           bf16* __restrict__ adjb) {
  int b = blockIdx.x >> 10, n = blockIdx.x & (NN - 1);
  const float* Srow  = S  + ((size_t)b << 20) + (size_t)n * NN;
  const float* STrow = ST + ((size_t)b << 20) + (size_t)n * NN;
  __shared__ float v[NN];
  __shared__ float red[256];
  float lsum = 0.f;
  for (int m = threadIdx.x; m < NN; m += 256) {
    float s = Srow[m] + STrow[m];                      // S[n,m] + S[m,n]
    float t = tanhf(s) * (1.f / F_TEMP);
    float e = __expf(t);
    v[m] = e; lsum += e;
  }
  red[threadIdx.x] = lsum;
  __syncthreads();
  for (int s = 128; s > 0; s >>= 1) {
    if (threadIdx.x < (unsigned)s) red[threadIdx.x] += red[threadIdx.x + s];
    __syncthreads();
  }
  float inv = 1.f / red[0];
  float* arow = adj_out + (((size_t)b << 10) + n) * NN;
  bf16*  brow = adjb    + (((size_t)b << 10) + n) * NN;
  for (int m = threadIdx.x; m < NN; m += 256) {
    float a = v[m] * inv;
    arow[m] = a;                       // f32 output (2nd tuple element)
    brow[m] = (bf16)(0.5f * a);        // 0.5*adj for na=(adj+I)/2
  }
}

// ---------------- 5) TCN (LDS-resident, WMMA) -------------------------------
struct TcnParams {
  const float* w1[6]; const float* b1[6];
  const float* w2[6]; const float* b2[6];
  const float* rw;    const float* rb;
};

static __device__ __forceinline__ v16bf
load_wfrag(const float* __restrict__ w, int IC, int k, int octile, int lane) {
  int oc    = octile * 16 + (lane & 15);
  int khalf = (lane & 16) ? 16 : 0;
  v16bf a;
#pragma unroll
  for (int i = 0; i < 8; ++i) {
    int ic0 = khalf + i;
    int ic1 = khalf + 8 + i;
    float v0 = (ic0 < IC) ? w[(oc * IC + ic0) * 3 + k] : 0.f;
    float v1 = (ic1 < IC) ? w[(oc * IC + ic1) * 3 + k] : 0.f;
    a[i] = (bf16)v0; a[8 + i] = (bf16)v1;
  }
  return a;
}

static __device__ __forceinline__ v16bf
load_hfrag(const bf16* __restrict__ hT, int ttile, int shift, int lane) {
  int t    = ttile * 16 + (lane & 15);
  int ksec = (lane & 16) ? 16 : 0;
  int ts   = t - shift;                 // causal left-pad with zeros
  v16bf b;
  if (ts >= 0) {
    const bf16* p = hT + ts * NDC + ksec;
#pragma unroll
    for (int i = 0; i < 16; ++i) b[i] = p[i];
  } else {
#pragma unroll
    for (int i = 0; i < 16; ++i) b[i] = (bf16)0.f;
  }
  return b;
}

__global__ __launch_bounds__(256)
void tcn_kernel(const float* __restrict__ x,
                const float* __restrict__ ne,
                TcnParams P,
                float* __restrict__ xt,
                bf16* __restrict__ zbT) {
  __shared__ bf16  hA[NL * NDC];        // [t][ch], time-major
  __shared__ bf16  hB[NL * NDC];
  __shared__ float fbuf[NL * NDC];
  __shared__ float gbuf[NL * NDC];

  int bn = blockIdx.x;
  int b  = bn >> 10, n = bn & (NN - 1);
  int tid = threadIdx.x;

  // Build padded input xc: ch 0..2 = x, 3..12 = node_emb, 13..31 = 0
  for (int idx = tid; idx < NL * NDC; idx += 256) {
    int t = idx >> 5, c = idx & 31;
    float v = 0.f;
    if (c < 3)       v = x[(((size_t)b * NL + t) * NN + n) * 3 + c];
    else if (c < NC) v = ne[n * 10 + (c - 3)];
    hA[idx] = (bf16)v;
  }
  __syncthreads();

  bf16* hin = hA;
  bf16* hout = hB;
  int dil = 1;
  int wave = tid >> 5, lane = tid & 31;

  for (int layer = 0; layer < 6; ++layer) {
    int IC = (layer == 0) ? NC : NDC;
    // 24 jobs: {f,g} x 2 oc-tiles x 6 t-tiles; 3 per wave
#pragma unroll
    for (int jj = 0; jj < 3; ++jj) {
      int job   = wave + jj * 8;
      int which = job / 12;
      int rem   = job % 12;
      int octile = rem / 6, ttile = rem % 6;
      const float* W = which ? P.w2[layer] : P.w1[layer];
      v8f acc = {};
#pragma unroll
      for (int k = 0; k < 3; ++k) {
        v16bf a   = load_wfrag(W, IC, k, octile, lane);
        v16bf bfr = load_hfrag(hin, ttile, (2 - k) * dil, lane);
        acc = wmma_bf16(a, bfr, acc);
      }
      float* obuf = which ? gbuf : fbuf;
      int t   = ttile * 16 + (lane & 15);
      int oc0 = octile * 16 + ((lane & 16) ? 8 : 0);
#pragma unroll
      for (int j = 0; j < 8; ++j) obuf[t * NDC + oc0 + j] = acc[j];
    }
    __syncthreads();

    // gated combine + residual
    const float* B1 = P.b1[layer];
    const float* B2 = P.b2[layer];
    for (int idx = tid; idx < NL * NDC; idx += 256) {
      int t = idx >> 5, oc = idx & 31;
      float f = tanhf(fbuf[idx] + B1[oc]);
      float g = gbuf[idx] + B2[oc];
      g = 1.f / (1.f + __expf(-g));
      float res;
      if (layer == 0) {
        float s = P.rb[oc];
#pragma unroll
        for (int ic = 0; ic < NC; ++ic)
          s += P.rw[oc * NC + ic] * (float)hin[t * NDC + ic];
        res = s;
      } else {
        res = (float)hin[idx];
      }
      float val = f * g + res;
      hout[idx] = (bf16)val;
      if (layer == 5 && t == NL - 1) {
        xt[(size_t)bn * NDC + oc] = val;                       // x_tcn (= z0)
        zbT[((size_t)b * NDC + oc) * NN + n] = (bf16)val;      // ch-major bf16
      }
    }
    __syncthreads();
    bf16* tmp = hin; hin = hout; hout = tmp;
    dil <<= 1;
  }
}

// ---------------- 6) APPNP step (WMMA, z panel in LDS) ----------------------
// z' = alpha*xt + (1-alpha)*( (0.5*adj)@z + 0.5*z )
// Block = 8 ntiles of one batch. Whole z panel [32][1024] bf16 (64KB) staged
// to LDS via TDM (preferred) / async-LDS / vector copy. Each wave computes
// BOTH c-tiles reusing one A-frag -> adj streamed from HBM exactly once.
__global__ __launch_bounds__(256)
void appnp_kernel(const bf16* __restrict__ adjb,
                  const bf16* __restrict__ zb_in,
                  const float* __restrict__ z_old,
                  const float* __restrict__ xt,
                  float* __restrict__ z_out,
                  bf16* __restrict__ zb_out) {
  __shared__ __align__(16) bf16 zsh[NDC * NN];          // 64 KB
  int tid  = threadIdx.x;
  int wave = tid >> 5, lane = tid & 31;
  int blk  = blockIdx.x;
  int b    = blk >> 3;                                  // 8 blocks per batch
  int ntile = ((blk & 7) << 3) | wave;
  const bf16* zg = zb_in + (size_t)b * NDC * NN;

#if USE_TDM
  if (wave == 0) {
    // 1-D TDM descriptor: 32K bf16 elements, tile == tensor (ISA ch.8 D#)
    unsigned long long ga = (unsigned long long)(uintptr_t)zg;
    unsigned lds = (unsigned)(uintptr_t)(void*)zsh;     // LDS byte address
    const unsigned TOT = NDC * NN;                      // 32768 elements
    v4u g0;
    g0[0] = 1u;                                         // count=1, user desc
    g0[1] = lds;                                        // lds_addr
    g0[2] = (unsigned)(ga & 0xFFFFFFFFu);               // global_addr[31:0]
    g0[3] = (unsigned)((ga >> 32) & 0x01FFFFFFu) | (2u << 30);  // type=2
    v8i g1;
    g1[0] = (int)(1u << 16);                            // data_size=1 (2B)
    g1[1] = (int)((TOT & 0xFFFFu) << 16);               // tensor_dim0 lo16
    g1[2] = (int)(((TOT >> 16) & 0xFFFFu) | (1u << 16)); // dim0 hi16, dim1=1
    g1[3] = (int)((TOT & 0xFFFFu) << 16);               // tile_dim0
    g1[4] = 0;                                          // tile_dim1/2 unused
    g1[5] = (int)TOT;                                   // dim0_stride lo32
    g1[6] = 0; g1[7] = 0;
    v4i g2 = {}; v4i g3 = {};
    v8i g4 = {};                                        // 6-arg form (clang-23)
    __builtin_amdgcn_tensor_load_to_lds(g0, g1, g2, g3, g4, 0);
#if __has_builtin(__builtin_amdgcn_s_wait_tensorcnt)
    __builtin_amdgcn_s_wait_tensorcnt(0);
#else
    asm volatile("s_wait_tensorcnt 0x0" ::: "memory");
#endif
  }
  __syncthreads();
#elif USE_ASYNC_LDS
  for (int i = tid; i < (NDC * NN) / 8; i += 256)       // 16B chunks
    __builtin_amdgcn_global_load_async_to_lds_b128(
        (as1void*)(zg + (size_t)i * 8), (as3void*)(zsh + i * 8), 0, 0);
#if __has_builtin(__builtin_amdgcn_s_wait_asynccnt)
  __builtin_amdgcn_s_wait_asynccnt(0);
#else
  asm volatile("s_wait_asynccnt 0x0" ::: "memory");
#endif
  __syncthreads();
#else
  for (int i = tid; i < (NDC * NN) / 8; i += 256)
    *(int4*)(zsh + i * 8) = *(const int4*)(zg + (size_t)i * 8);
  __syncthreads();
#endif

  int nrow  = ntile * 16 + (lane & 15);
  int cl    = lane & 15;
  int khalf = (lane & 16) ? 8 : 0;
  int ksec  = (lane & 16) ? 16 : 0;
  const bf16* arow = adjb + (((size_t)b << 10) + nrow) * NN;
  const bf16* z0sh = zsh + (size_t)cl * NN + ksec;           // c-tile 0
  const bf16* z1sh = zsh + (size_t)(16 + cl) * NN + ksec;    // c-tile 1
  v8f acc0 = {}, acc1 = {};
  for (int kc = 0; kc < 32; ++kc) {
    int kb = kc * 32;
    if (kc + 8 < 32) __builtin_prefetch(arow + kb + 256, 0, 0);
    v16bf a, b0, b1;
    const bf16* ap = arow + kb + ksec;
#pragma unroll
    for (int i = 0; i < 8; ++i) { a[i] = ap[i]; a[8 + i] = ap[8 + i]; }
#pragma unroll
    for (int i = 0; i < 16; ++i) { b0[i] = z0sh[kb + i]; b1[i] = z1sh[kb + i]; }
    acc0 = wmma_bf16(a, b0, acc0);       // same A reused for both c-tiles
    acc1 = wmma_bf16(a, b1, acc1);
  }
  int n0 = ntile * 16 + ((lane & 16) ? 8 : 0);
#pragma unroll
  for (int ct = 0; ct < 2; ++ct) {
    int c = ct * 16 + cl;
    v8f acc = ct ? acc1 : acc0;
#pragma unroll
    for (int j = 0; j < 8; ++j) {
      int nn = n0 + j;
      size_t zi = (((size_t)b << 10) + nn) * NDC + c;
      float znew = F_ALPHA * xt[zi] +
                   (1.f - F_ALPHA) * (acc[j] + 0.5f * z_old[zi]);
      z_out[zi] = znew;
      zb_out[((size_t)b * NDC + c) * NN + nn] = (bf16)znew;
    }
  }
}

// ---------------- 7) head MLP -----------------------------------------------
__global__ void head_kernel(const float* __restrict__ zf,
                            const float* __restrict__ xt,
                            const float* __restrict__ hw1, const float* __restrict__ hb1,
                            const float* __restrict__ hw2, const float* __restrict__ hb2,
                            float* __restrict__ out) {
  int g    = threadIdx.x >> 5;                 // 4 nodes / block of 128
  int lane = threadIdx.x & 31;
  int node = blockIdx.x * 4 + g;
  __shared__ float zrow[4][NDC];
  __shared__ float hid[4][NDC];
  float z  = zf[(size_t)node * NDC + lane];
  float xv = xt[(size_t)node * NDC + lane];
  zrow[g][lane] = fmaxf(z, 0.f) + xv;          // relu(z) + x_tcn
  __syncthreads();
  float h = hb1[lane];
#pragma unroll
  for (int i = 0; i < NDC; ++i) h += zrow[g][i] * hw1[i * NDC + lane];
  hid[g][lane] = fmaxf(h, 0.f);
  __syncthreads();
  if (lane < NOUT) {
    float o = hb2[lane];
#pragma unroll
    for (int i = 0; i < NDC; ++i) o += hid[g][i] * hw2[i * NOUT + lane];
    out[(size_t)node * NOUT + lane] = o;
  }
}

// ---------------------------------------------------------------------------
extern "C" void kernel_launch(void* const* d_in, const int* in_sizes, int n_in,
                              void* d_out, int out_size, void* d_ws, size_t ws_size,
                              hipStream_t stream) {
  (void)in_sizes; (void)n_in; (void)out_size; (void)ws_size;
  const float* x   = (const float*)d_in[0];
  const float* ne  = (const float*)d_in[1];
  const float* dw1 = (const float*)d_in[2];
  const float* db1 = (const float*)d_in[3];
  const float* dw2 = (const float*)d_in[4];
  const float* db2 = (const float*)d_in[5];
  const float* de1 = (const float*)d_in[6];
  const float* de2 = (const float*)d_in[7];
  TcnParams P;
  P.w1[0] = (const float*)d_in[8];  P.b1[0] = (const float*)d_in[9];
  P.w2[0] = (const float*)d_in[10]; P.b2[0] = (const float*)d_in[11];
  P.rw    = (const float*)d_in[12]; P.rb    = (const float*)d_in[13];
  for (int l = 1; l < 6; ++l) {
    int base = 14 + (l - 1) * 4;
    P.w1[l] = (const float*)d_in[base + 0];
    P.b1[l] = (const float*)d_in[base + 1];
    P.w2[l] = (const float*)d_in[base + 2];
    P.b2[l] = (const float*)d_in[base + 3];
  }
  const float* hw1 = (const float*)d_in[34];
  const float* hb1 = (const float*)d_in[35];
  const float* hw2 = (const float*)d_in[36];
  const float* hb2 = (const float*)d_in[37];

  float* out_head = (float*)d_out;                       // [16,1024,12]
  float* out_adj  = (float*)d_out + (size_t)NB * NN * NOUT;

  // Workspace layout (1 MiB-aligned regions)
  char* w = (char*)d_ws;
  const size_t MB = 1u << 20;
  float* xadj = (float*)(w + 0 * MB);     // 16*1024*16 f32        = 1 MB
  bf16*  m1b  = (bf16*)(w + 1 * MB);      // 16*1024*64 bf16       = 2 MB
  bf16*  m2b  = (bf16*)(w + 3 * MB);      //                         2 MB
  float* S    = (float*)(w + 5 * MB);     // 16*1024*1024 f32      = 64 MB
  bf16*  adjb = (bf16*)(w + 69 * MB);     // 0.5*adj bf16          = 32 MB
  float* xt   = (float*)(w + 101 * MB);   // 16*1024*32 f32        = 2 MB
  float* zfA  = (float*)(w + 103 * MB);   // 2 MB
  float* zfB  = (float*)(w + 105 * MB);   // 2 MB
  bf16*  zbA  = (bf16*)(w + 107 * MB);    // [b][32][1024] bf16    = 1 MB
  bf16*  zbB  = (bf16*)(w + 108 * MB);    // 1 MB
  float* ST   = (float*)(w + 109 * MB);   // S transposed f32      = 64 MB

  xadj_kernel<<<dim3((NB * NN + 255) / 256), dim3(256), 0, stream>>>(x, ne, xadj);
  ddl_kernel<<<dim3(NB * NN / 4), dim3(256), 0, stream>>>(
      xadj, dw1, db1, dw2, db2, de1, de2, m1b, m2b);
  sgemm_kernel<<<dim3(NB * 64 * 64 / 8), dim3(256), 0, stream>>>(m1b, m2b, S, ST);
  adj_kernel<<<dim3(NB * NN), dim3(256), 0, stream>>>(S, ST, out_adj, adjb);
  tcn_kernel<<<dim3(NB * NN), dim3(256), 0, stream>>>(x, ne, P, xt, zbA);

  const bf16* zin = zbA; bf16* zout = zbB;
  const float* zold = xt; float* zf_out = zfA;
  for (int i = 0; i < 8; ++i) {
    appnp_kernel<<<dim3(NB * 8), dim3(256), 0, stream>>>(
        adjb, zin, zold, xt, zf_out, zout);
    const bf16* t = zin; zin = zout; zout = (bf16*)t;
    zold = zf_out; zf_out = (zf_out == zfA) ? zfB : zfA;
  }
  head_kernel<<<dim3(NB * NN / 4), dim3(128), 0, stream>>>(
      zold, xt, hw1, hb1, hw2, hb2, out_head);
}
